// ClassicalSelfAttention_65481071406199
// MI455X (gfx1250) — matrix operand order repaired and checked by reading, not compiled, not run on hardware
//
#include <hip/hip_runtime.h>
#include <hip/hip_bf16.h>

#define EMBED   512
#define NHEADS  8
#define HDIM    64
#define SEQ     4096
#define BATCH   2
#define ROWS    (BATCH*SEQ)     /* 8192 */
#define ATT_SCALE 0.125f        /* 1/sqrt(64) */
#define LN_EPS  1e-5f

typedef __attribute__((ext_vector_type(16))) _Float16 v16h;
typedef __attribute__((ext_vector_type(8)))  _Float16 v8h;
typedef __attribute__((ext_vector_type(8)))  float    v8f;

__device__ __forceinline__ v8f wmma_f16(v16h a, v16h b, v8f c) {
  // D = A(16x32 f16) * B(32x16 f16) + C(16x16 f32)
  return __builtin_amdgcn_wmma_f32_16x16x32_f16(false, a, false, b, (short)0, c,
                                                false, false);
}

// A fragment 16x32 from row-major [M][K] matrix; rows m0..m0+15, cols k0..k0+31.
// lane L: M = m0 + (L&15); element e -> K = k0 + e + (L>>4)*16
__device__ __forceinline__ v16h load_A_rm(const _Float16* base, int ld, int m0,
                                          int k0, int lane) {
  const int hf = lane >> 4;
  const _Float16* p = base + (size_t)(m0 + (lane & 15)) * ld + k0 + hf * 16;
  v8h lo = *(const v8h*)(p);
  v8h hi = *(const v8h*)(p + 8);
  v16h a;
#pragma unroll
  for (int i = 0; i < 8; ++i) { a[i] = lo[i]; a[i + 8] = hi[i]; }
  return a;
}

// B fragment 32x16 where B[k][n] = Mrm[n][k], Mrm row-major [N][K] (e.g. torch
// Linear weight [out,in]).  lane L: N = n0 + (L&15); element e -> K = k0 + e + (L>>4)*16
__device__ __forceinline__ v16h load_B_t(const _Float16* Mrm, int ld, int k0,
                                         int n0, int lane) {
  const int hf = lane >> 4;
  return *(const v16h*)(Mrm + (size_t)(n0 + (lane & 15)) * ld + k0 + hf * 16);
}

// ---------------------------------------------------------------- convert f32->f16
__global__ void cast_f16_kernel(const float* __restrict__ src,
                                _Float16* __restrict__ dst, int n) {
  int i = blockIdx.x * blockDim.x + threadIdx.x;
  if (i < n) dst[i] = (_Float16)src[i];
}

// ---------------------------------------------------------------- QKV projection
// Y = X @ W^T for W in {Wq,Wk,Wv} (mode = blockIdx.z). Q,K stored [b,h,s,dh];
// V stored transposed [b,h,dh,s] so PV's B-operand is contiguous.
__global__ __launch_bounds__(128) void qkv_gemm_kernel(
    const _Float16* __restrict__ Xh,  const _Float16* __restrict__ Wqh,
    const _Float16* __restrict__ Wkh, const _Float16* __restrict__ Wvh,
    _Float16* __restrict__ Qh, _Float16* __restrict__ Kh,
    _Float16* __restrict__ Vt) {
  const int lane = threadIdx.x & 31;
  const int wave = threadIdx.x >> 5;
  const int m0   = blockIdx.x * 16;
  const int n0   = blockIdx.y * 64 + wave * 16;
  const int mode = blockIdx.z;
  const _Float16* W = (mode == 0) ? Wqh : (mode == 1) ? Wkh : Wvh;

  v8f acc = {};
#pragma unroll 4
  for (int k = 0; k < EMBED; k += 32) {
    v16h a = load_A_rm(Xh, EMBED, m0, k, lane);
    v16h bf = load_B_t(W, EMBED, k, n0, lane);
    acc = wmma_f16(a, bf, acc);
  }

  const int hf  = lane >> 4;
  const int col = n0 + (lane & 15);
  const int hd  = col >> 6;        // head
  const int dh  = col & 63;        // dim within head
#pragma unroll
  for (int r = 0; r < 8; ++r) {
    int m = m0 + r + hf * 8;
    int bb = m / SEQ, ss = m % SEQ;
    size_t bh = (size_t)(bb * NHEADS + hd);
    _Float16 val = (_Float16)acc[r];
    if (mode == 0)      Qh[(bh * SEQ + ss) * HDIM + dh] = val;
    else if (mode == 1) Kh[(bh * SEQ + ss) * HDIM + dh] = val;
    else                Vt[(bh * HDIM + dh) * SEQ + ss] = val;
  }
}

// ---------------------------------------------------------------- flash attention
// Block = 4 waves; all waves share one (b,h) and the same 32-key window, staged
// through LDS once per block (4x L2 traffic reduction). Wave w owns Q rows
// [blockIdx.x*64 + w*16, +16). Online softmax, WMMA for QK^T and PV.
__global__ __launch_bounds__(128) void flash_attn_kernel(
    const _Float16* __restrict__ Qh, const _Float16* __restrict__ Kh,
    const _Float16* __restrict__ Vt, _Float16* __restrict__ Oh) {
  __shared__ __align__(32) _Float16 Ktile[32 * HDIM];   // [t_local][d]   4KB
  __shared__ __align__(32) _Float16 Vtile[HDIM * 32];   // [d][t_local]   4KB
  __shared__ __align__(32) _Float16 pls[4][16 * 32];    // per-wave P     4KB
  const int tid  = threadIdx.x;
  const int lane = tid & 31;
  const int wave = tid >> 5;
  const int bh = blockIdx.y;                 // 0..15
  const int bb = bh >> 3, hh = bh & 7;
  const int q0 = blockIdx.x * 64 + wave * 16;
  const _Float16* Q = Qh + (size_t)bh * SEQ * HDIM;
  const _Float16* K = Kh + (size_t)bh * SEQ * HDIM;
  const _Float16* V = Vt + (size_t)bh * HDIM * SEQ;    // [d][t]
  _Float16* pbuf = pls[wave];
  const int hf = lane >> 4;
  const int ln16 = lane & 15;

  // cooperative-copy coordinates (32B per thread per tile)
  const int krow = tid >> 2, kcb = tid & 3;   // K: 32 rows x 4 chunks of 16h
  const int vrow = tid >> 1, vcb = tid & 1;   // V: 64 rows x 2 chunks of 16h

  // Q tile (16x64) held in registers as two A fragments
  const v16h aq0 = load_A_rm(Q, HDIM, q0, 0, lane);
  const v16h aq1 = load_A_rm(Q, HDIM, q0, 32, lane);

  v8f acc0 = {}, acc1 = {}, acc2 = {}, acc3 = {};      // 16x64 output accum
  float mrow[8], lrow[8];
#pragma unroll
  for (int r = 0; r < 8; ++r) { mrow[r] = -3.0e38f; lrow[r] = 0.0f; }

  for (int t0 = 0; t0 < SEQ; t0 += 32) {
    // ---- stage K/V tiles into LDS (whole block cooperates) ----
    __syncthreads();   // previous iteration's compute done before overwrite
    *(v16h*)(Ktile + krow * HDIM + kcb * 16) =
        *(const v16h*)(K + (size_t)(t0 + krow) * HDIM + kcb * 16);
    *(v16h*)(Vtile + vrow * 32 + vcb * 16) =
        *(const v16h*)(V + (size_t)vrow * SEQ + t0 + vcb * 16);
    if (t0 + 32 < SEQ) {
      __builtin_prefetch(K + (size_t)(t0 + 32 + krow) * HDIM, 0, 0);
      __builtin_prefetch(V + (size_t)vrow * SEQ + t0 + 32, 0, 0);
    }
    __syncthreads();

    // ---- scores: two 16x16 tiles (keys t0..t0+15 and t0+16..t0+31) ----
    v8f c0 = {}, c1 = {};
    c0 = wmma_f16(aq0, load_B_t(Ktile, HDIM, 0,  0,  lane), c0);
    c0 = wmma_f16(aq1, load_B_t(Ktile, HDIM, 32, 0,  lane), c0);
    c1 = wmma_f16(aq0, load_B_t(Ktile, HDIM, 0,  16, lane), c1);
    c1 = wmma_f16(aq1, load_B_t(Ktile, HDIM, 32, 16, lane), c1);

    float corr[8];
#pragma unroll
    for (int r = 0; r < 8; ++r) {
      float s0 = c0[r] * ATT_SCALE;
      float s1 = c1[r] * ATT_SCALE;
      float rm = fmaxf(s0, s1);                        // row-max over 16 lanes
      rm = fmaxf(rm, __shfl_xor(rm, 1));
      rm = fmaxf(rm, __shfl_xor(rm, 2));
      rm = fmaxf(rm, __shfl_xor(rm, 4));
      rm = fmaxf(rm, __shfl_xor(rm, 8));
      float mn = fmaxf(mrow[r], rm);
      corr[r] = __expf(mrow[r] - mn);
      float p0 = __expf(s0 - mn);
      float p1 = __expf(s1 - mn);
      c0[r] = p0; c1[r] = p1;
      float rs = p0 + p1;
      rs += __shfl_xor(rs, 1);
      rs += __shfl_xor(rs, 2);
      rs += __shfl_xor(rs, 4);
      rs += __shfl_xor(rs, 8);
      lrow[r] = lrow[r] * corr[r] + rs;
      mrow[r] = mn;
    }
#pragma unroll
    for (int r = 0; r < 8; ++r) {
      acc0[r] *= corr[r]; acc1[r] *= corr[r];
      acc2[r] *= corr[r]; acc3[r] *= corr[r];
    }
    // ---- P tile (16x32) C-layout -> LDS -> reload in A-layout ----
#pragma unroll
    for (int r = 0; r < 8; ++r) {
      int row = r + hf * 8;
      pbuf[row * 32 + ln16]      = (_Float16)c0[r];
      pbuf[row * 32 + 16 + ln16] = (_Float16)c1[r];
    }
    asm volatile("s_wait_dscnt 0" ::: "memory");   // in-wave RAW across lanes
    v16h pa = load_A_rm(pbuf, 32, 0, 0, lane);
    // ---- PV: B[k=t_local][n=d] = Vtile[d][t_local] (contiguous per lane) ----
    acc0 = wmma_f16(pa, load_B_t(Vtile, 32, 0, 0,  lane), acc0);
    acc1 = wmma_f16(pa, load_B_t(Vtile, 32, 0, 16, lane), acc1);
    acc2 = wmma_f16(pa, load_B_t(Vtile, 32, 0, 32, lane), acc2);
    acc3 = wmma_f16(pa, load_B_t(Vtile, 32, 0, 48, lane), acc3);
    asm volatile("" ::: "memory");   // keep next iter's ds stores behind the loads
  }

#pragma unroll
  for (int r = 0; r < 8; ++r) {
    float inv = 1.0f / lrow[r];
    int row = q0 + r + hf * 8;
    size_t base = ((size_t)bb * SEQ + row) * EMBED + hh * 64 + ln16;
    Oh[base + 0]  = (_Float16)(acc0[r] * inv);
    Oh[base + 16] = (_Float16)(acc1[r] * inv);
    Oh[base + 32] = (_Float16)(acc2[r] * inv);
    Oh[base + 48] = (_Float16)(acc3[r] * inv);
  }
}

// ---------------------------------------------------------------- out proj + residual
__global__ __launch_bounds__(128) void out_proj_kernel(
    const _Float16* __restrict__ Oh, const _Float16* __restrict__ Woh,
    const float* __restrict__ Xin, float* __restrict__ Y) {
  const int lane = threadIdx.x & 31;
  const int wave = threadIdx.x >> 5;
  const int m0 = blockIdx.x * 16;
  const int n0 = blockIdx.y * 64 + wave * 16;
  v8f acc = {};
#pragma unroll 4
  for (int k = 0; k < EMBED; k += 32) {
    v16h a  = load_A_rm(Oh, EMBED, m0, k, lane);
    v16h bf = load_B_t(Woh, EMBED, k, n0, lane);
    acc = wmma_f16(a, bf, acc);
  }
  const int hf = lane >> 4;
  const int col = n0 + (lane & 15);
#pragma unroll
  for (int r = 0; r < 8; ++r) {
    size_t idx = (size_t)(m0 + r + hf * 8) * EMBED + col;
    Y[idx] = acc[r] + Xin[idx];          // residual add
  }
}

// ---------------------------------------------------------------- layernorm (512 cols)
__global__ __launch_bounds__(256) void layernorm_kernel(
    const float* __restrict__ Y, const float* __restrict__ gamma,
    const float* __restrict__ beta, float* __restrict__ out) {
  __shared__ float red[256];
  const int t = threadIdx.x;
  const float* y = Y + (size_t)blockIdx.x * EMBED;
  float x0 = y[t], x1 = y[t + 256];
  red[t] = x0 + x1;
  __syncthreads();
  for (int off = 128; off > 0; off >>= 1) {
    if (t < off) red[t] += red[t + off];
    __syncthreads();
  }
  float mu = red[0] * (1.0f / EMBED);
  __syncthreads();
  float d0 = x0 - mu, d1 = x1 - mu;
  red[t] = d0 * d0 + d1 * d1;
  __syncthreads();
  for (int off = 128; off > 0; off >>= 1) {
    if (t < off) red[t] += red[t + off];
    __syncthreads();
  }
  float rstd = rsqrtf(red[0] * (1.0f / EMBED) + LN_EPS);
  float* o = out + (size_t)blockIdx.x * EMBED;
  o[t]       = d0 * rstd * gamma[t]       + beta[t];
  o[t + 256] = d1 * rstd * gamma[t + 256] + beta[t + 256];
}

// ---------------------------------------------------------------- launcher
extern "C" void kernel_launch(void* const* d_in, const int* in_sizes, int n_in,
                              void* d_out, int out_size, void* d_ws, size_t ws_size,
                              hipStream_t stream) {
  (void)in_sizes; (void)n_in; (void)out_size; (void)ws_size;
  const float* Xin = (const float*)d_in[2];
  const float* Wq  = (const float*)d_in[3];
  const float* Wk  = (const float*)d_in[4];
  const float* Wv  = (const float*)d_in[5];
  const float* Wo  = (const float*)d_in[6];
  const float* ga  = (const float*)d_in[7];
  const float* be  = (const float*)d_in[8];
  float* Out = (float*)d_out;

  char* ws = (char*)d_ws;
  size_t off = 0;
  auto carve = [&](size_t bytes) {
    void* p = ws + off;
    off += (bytes + 255) & ~(size_t)255;
    return p;
  };
  const size_t actN = (size_t)ROWS * EMBED;   // 8192*512
  const size_t wN   = (size_t)EMBED * EMBED;  // 512*512
  _Float16* Xh  = (_Float16*)carve(actN * 2);
  _Float16* Wqh = (_Float16*)carve(wN * 2);
  _Float16* Wkh = (_Float16*)carve(wN * 2);
  _Float16* Wvh = (_Float16*)carve(wN * 2);
  _Float16* Woh = (_Float16*)carve(wN * 2);
  _Float16* Qh  = (_Float16*)carve(actN * 2);
  _Float16* Kh  = (_Float16*)carve(actN * 2);
  _Float16* Vt  = (_Float16*)carve(actN * 2);
  _Float16* Oh  = (_Float16*)carve(actN * 2);
  float*    Yf  = (float*)   carve(actN * 4);

  // 1. cast activations + weights to f16
  cast_f16_kernel<<<(int)((actN + 255) / 256), 256, 0, stream>>>(Xin, Xh, (int)actN);
  cast_f16_kernel<<<(int)((wN + 255) / 256), 256, 0, stream>>>(Wq, Wqh, (int)wN);
  cast_f16_kernel<<<(int)((wN + 255) / 256), 256, 0, stream>>>(Wk, Wkh, (int)wN);
  cast_f16_kernel<<<(int)((wN + 255) / 256), 256, 0, stream>>>(Wv, Wvh, (int)wN);
  cast_f16_kernel<<<(int)((wN + 255) / 256), 256, 0, stream>>>(Wo, Woh, (int)wN);

  // 2. QKV projections (WMMA GEMM), V stored transposed
  qkv_gemm_kernel<<<dim3(ROWS / 16, EMBED / 64, 3), 128, 0, stream>>>(
      Xh, Wqh, Wkh, Wvh, Qh, Kh, Vt);

  // 3. fused flash attention (LDS-shared K/V tiles)
  flash_attn_kernel<<<dim3(SEQ / 64, BATCH * NHEADS), 128, 0, stream>>>(Qh, Kh, Vt, Oh);

  // 4. output projection + residual
  out_proj_kernel<<<dim3(ROWS / 16, EMBED / 64), 128, 0, stream>>>(Oh, Woh, Xin, Yf);

  // 5. layernorm
  layernorm_kernel<<<ROWS, 256, 0, stream>>>(Yf, ga, be, Out);
}
